// Dynamic_7x7_naivev2_40338332844079
// MI455X (gfx1250) — matrix operand order repaired and checked by reading, not compiled, not run on hardware
//
#include <hip/hip_runtime.h>
#include <hip/hip_bf16.h>
#include <stddef.h>

// ---------------------------------------------------------------------------
// Problem constants (reference: B=4, H=480, W=640, 48 guidance ch, 24 dyn ch)
// ---------------------------------------------------------------------------
#define BB   4
#define HH   480
#define WW   640
#define CG   48
#define HWSZ (HH * WW)
#define PLANE (BB * HWSZ)   // elements in one (B,1,H,W) plane

typedef __attribute__((ext_vector_type(16))) __bf16 v16bf;
typedef __attribute__((ext_vector_type(8)))  float  v8f;
typedef __attribute__((ext_vector_type(4)))  unsigned int u32x4;
typedef __attribute__((ext_vector_type(4)))  int  i32x4;
typedef __attribute__((ext_vector_type(8)))  int  i32x8;

union V16BF { v16bf v; unsigned int u[8]; };

// ---------------------------------------------------------------------------
// Per-channel tap offsets (see round-0 derivation from make_edge_weight()).
// ---------------------------------------------------------------------------
__device__ __constant__ signed char OFF_DY[CG] = {
   1, 1, 1, 0, 0,-1,-1,-1,
   2, 2, 2, 2, 2,  1, 1,  0, 0, -1,-1, -2,-2,-2,-2,-2,
   3, 3, 3, 3, 3, 3, 3,  2, 2,  1, 1,  0, 0, -1,-1, -2,-2, -3,-3,-3,-3,-3,-3,-3
};
__device__ __constant__ signed char OFF_DX[CG] = {
   1, 0,-1, 1,-1, 1, 0,-1,
   2, 1, 0,-1,-2,  2,-2,  2,-2,  2,-2,  2, 1, 0,-1,-2,
   3, 2, 1, 0,-1,-2,-3,  3,-3,  3,-3,  3,-3,  3,-3,  3,-3,  3, 2, 1, 0,-1,-2,-3
};

// ---------------------------------------------------------------------------
// helpers
// ---------------------------------------------------------------------------
__device__ __forceinline__ unsigned short f2bf(float f) {
  union { float f; unsigned int i; } cv; cv.f = f;
  unsigned int x = cv.i;
  x += 0x7FFFu + ((x >> 16) & 1u);          // round-to-nearest-even
  return (unsigned short)(x >> 16);
}
__device__ __forceinline__ float bf2f(unsigned short u) {
  union { unsigned int i; float f; } cv; cv.i = ((unsigned int)u) << 16;
  return cv.f;
}
__device__ __forceinline__ float sigmoidf(float x) {
  return 1.0f / (1.0f + __expf(-x));
}

// ---------------------------------------------------------------------------
// Kernel 1: iteration-invariant precompute.
//   g_bf16[c] = bf16(guidance[c]); aff[0..5] = ring sums of g and |g|;
//   fixp = sign(feat_fix) * confidence
// ---------------------------------------------------------------------------
__global__ void precompute_kernel(const float* __restrict__ guid,
                                  const float* __restrict__ conf,
                                  const float* __restrict__ ffix,
                                  unsigned short* __restrict__ gbf,
                                  float* __restrict__ aff,
                                  float* __restrict__ fixp) {
  int idx = blockIdx.x * blockDim.x + threadIdx.x;
  if (idx >= PLANE) return;
  int b = idx / HWSZ;
  int p = idx - b * HWSZ;
  int y = p / WW;
  int x = p - y * WW;

  float as[3] = {0.f, 0.f, 0.f};
  float ab[3] = {0.f, 0.f, 0.f};

  #pragma unroll 4
  for (int c = 0; c < CG; ++c) {
    size_t base = ((size_t)(b * CG + c)) * HWSZ;
    float g = guid[base + (size_t)y * WW + x];
    gbf[base + (size_t)y * WW + x] = f2bf(g);

    int ny = y + OFF_DY[c];
    int nx = x + OFF_DX[c];
    float gn = 0.f;
    if ((unsigned)ny < (unsigned)HH && (unsigned)nx < (unsigned)WW)
      gn = guid[base + (size_t)ny * WW + nx];
    int r = (c < 8) ? 0 : ((c < 24) ? 1 : 2);
    as[r] += gn;
    ab[r] += fabsf(gn);
  }
  aff[0 * PLANE + idx] = as[0];
  aff[1 * PLANE + idx] = as[1];
  aff[2 * PLANE + idx] = as[2];
  aff[3 * PLANE + idx] = ab[0];
  aff[4 * PLANE + idx] = ab[1];
  aff[5 * PLANE + idx] = ab[2];

  float ff = ffix[idx];
  float sg = (float)(ff > 0.f) - (float)(ff < 0.f);
  fixp[idx] = sg * conf[idx];
}

// ---------------------------------------------------------------------------
// Per-wave WMMA rows: 16-pixel group per row; B[k,n] = bf16(g*f) gathered at
// per-channel offsets; A = constant ring-selection (rows 0..2 = s3,s5,s7).
// CHECK=false (interior tile): no bounds checks -> gathers fold into
// immediate-offset global_load_d16 off one per-lane base.
// ---------------------------------------------------------------------------
template <bool CHECK>
__device__ __forceinline__ void prop_rows(const unsigned short* __restrict__ gbf,
                                          const float* __restrict__ dynm,
                                          const float* __restrict__ finit,
                                          const float* __restrict__ ffix,
                                          const float* __restrict__ fixp,
                                          const float* __restrict__ aff,
                                          float* __restrict__ dst,
                                          const float* sF,
                                          int b, int tx0, int ty0, int iter) {
  const int tid  = threadIdx.x;
  const int wave = tid >> 5;
  const int lane = tid & 31;
  const int n    = lane & 15;   // pixel column (WMMA N), also A row index M
  const int half = lane >> 4;   // K-half held by this lane
  const int m    = n;

  // ---- constant ring-selection A matrices ---------------------------------
  // A k order here (contiguous, same as B): lane<16: v -> K=2v,2v+1 (K 0..15)
  //                          lane>=16: v -> K=16+2v,16+2v+1 (K 16..31)
  V16BF a0, a1;
  #pragma unroll
  for (int v = 0; v < 8; ++v) {
    int k0 = half * 16 + 2 * v;
    int k1 = k0 + 1;
    unsigned int p0 = 0, p1 = 0;
    if (m <= 2) {   // row 0 -> ring3 (k<8), row 1 -> ring5 (8..23), row 2 -> ring7
      bool s0 = (m == 0) ? (k0 < 8) : (m == 1) ? (k0 >= 8 && k0 < 24) : (k0 >= 24);
      bool s1 = (m == 0) ? (k1 < 8) : (m == 1) ? (k1 >= 8 && k1 < 24) : (k1 >= 24);
      p0 = s0 ? 0x3F80u : 0u;
      p1 = s1 ? 0x3F80u : 0u;
    }
    a0.u[v] = p0 | (p1 << 16);
    unsigned int q0 = 0, q1 = 0;
    if (m == 2) {   // s7 tail: channels 32..47 live at K 0..15 of tile 1
      q0 = (k0 < 16) ? 0x3F80u : 0u;
      q1 = (k1 < 16) ? 0x3F80u : 0u;
    }
    a1.u[v] = q0 | (q1 << 16);
  }

  const size_t gplane = (size_t)b * CG * HWSZ;

  for (int r = 0; r < 2; ++r) {
    const int ly = wave * 2 + r;      // row within tile, 0..15
    const int gy = ty0 + ly;
    const int gx = tx0 + n;

    // prefetch next row's guidance line into L2 (CDNA5 global_prefetch_b8)
    if (ly + 1 < 16)
      __builtin_prefetch(&gbf[gplane + (size_t)(ty0 + ly + 1) * WW + tx0], 0, 1);

    // per-lane base for interior gathers (all offsets become immediates)
    const unsigned short* gb = gbf + gplane + (size_t)gy * WW + gx;

    V16BF b0, b1;
    #pragma unroll
    for (int v = 0; v < 8; ++v) {
      // B k order here (contiguous, same as A): lane<16 holds K=0..15 (v -> 2v,2v+1); lane>=16 K=16..31
      int kb = half * 16 + 2 * v;
      unsigned int pk[2];
      #pragma unroll
      for (int s = 0; s < 2; ++s) {
        int c  = kb + s;                        // tile 0: channels 0..31
        int dy = OFF_DY[c], dx = OFF_DX[c];
        float g;
        if (CHECK) {
          int yy = gy + dy, xx = gx + dx;
          g = 0.f;
          if ((unsigned)yy < (unsigned)HH && (unsigned)xx < (unsigned)WW)
            g = bf2f(gbf[gplane + (size_t)c * HWSZ + (size_t)yy * WW + xx]);
        } else {
          g = bf2f(gb[(size_t)c * HWSZ + dy * WW + dx]);
        }
        float f = sF[(ly + 3 + dy) * 22 + (n + 3 + dx)];
        pk[s] = f2bf(g * f);
      }
      b0.u[v] = pk[0] | (pk[1] << 16);

      #pragma unroll
      for (int s = 0; s < 2; ++s) {
        int c = 32 + kb + s;                    // tile 1: channels 32..47
        unsigned int val = 0;
        if (c < CG) {
          int dy = OFF_DY[c], dx = OFF_DX[c];
          float g;
          if (CHECK) {
            int yy = gy + dy, xx = gx + dx;
            g = 0.f;
            if ((unsigned)yy < (unsigned)HH && (unsigned)xx < (unsigned)WW)
              g = bf2f(gbf[gplane + (size_t)c * HWSZ + (size_t)yy * WW + xx]);
          } else {
            g = bf2f(gb[(size_t)c * HWSZ + dy * WW + dx]);
          }
          float f = sF[(ly + 3 + dy) * 22 + (n + 3 + dx)];
          val = f2bf(g * f);
        }
        pk[s] = val;
      }
      b1.u[v] = pk[0] | (pk[1] << 16);
    }

    // ---- chained WMMAs: lane n gets s3,s5,s7 in acc[0..2] ------------------
    v8f acc = {};
    acc = __builtin_amdgcn_wmma_f32_16x16x32_bf16(false, a0.v, false, b0.v,
                                                  (short)0, acc, false, false);
    acc = __builtin_amdgcn_wmma_f32_16x16x32_bf16(false, a1.v, false, b1.v,
                                                  (short)0, acc, false, false);

    // ---- scalar epilogue: low lane half only (results live there) ----------
    if (lane < 16) {
      float s3 = acc[0], s5 = acc[1], s7 = acc[2];
      size_t pidx = (size_t)b * HWSZ + (size_t)gy * WW + gx;
      size_t drow = (size_t)gy * WW + gx;
      float att0 = sigmoidf(dynm[((size_t)(b * 24 + 4 * iter + 0)) * HWSZ + drow]);
      float att1 = sigmoidf(dynm[((size_t)(b * 24 + 4 * iter + 1)) * HWSZ + drow]);
      float att2 = sigmoidf(dynm[((size_t)(b * 24 + 4 * iter + 2)) * HWSZ + drow]);
      float att3 = sigmoidf(dynm[((size_t)(b * 24 + 4 * iter + 3)) * HWSZ + drow]);

      float as3 = aff[0 * PLANE + pidx], as5 = aff[1 * PLANE + pidx], as7 = aff[2 * PLANE + pidx];
      float ab3 = aff[3 * PLANE + pidx], ab5 = aff[4 * PLANE + pidx], ab7 = aff[5 * PLANE + pidx];

      float gate_abs = att0 * ab3 + att1 * ab5 + att2 * ab7 + att3 + 1e-4f;
      float gate     = att0 * as3 + att1 * as5 + att2 * as7 + att3;

      float fc = sF[(ly + 3) * 22 + (n + 3)];
      float fi = finit[pidx];
      float fr = (att0 * s3 + att1 * s5 + att2 * s7 + att3 * fc +
                  (gate_abs - gate) * fi) / gate_abs;
      float fx = fixp[pidx];
      fr = (1.0f - fx) * fr + fx * ffix[pidx];
      dst[pidx] = fr;
    }
  }
}

// ---------------------------------------------------------------------------
// Kernel 2: one propagation iteration.  Block = 256 thr, tile = 16x16 pixels.
// Interior tiles: feat tile (+3 halo) fetched via Tensor Data Mover
// (tensor_load_to_lds, 2D descriptor: 22x22 tile, stride W) and unchecked
// gathers.  Border tiles: zero-padded scalar fill + checked gathers.
// ---------------------------------------------------------------------------
__global__ void __launch_bounds__(256)
prop_iter_kernel(const unsigned short* __restrict__ gbf,
                 const float* __restrict__ dynm,
                 const float* __restrict__ finit,
                 const float* __restrict__ ffix,
                 const float* __restrict__ fixp,
                 const float* __restrict__ aff,
                 const float* __restrict__ src,
                 float* __restrict__ dst,
                 int iter) {
  __shared__ float sF[22 * 22];   // 16x16 tile + 3-halo

  const int b   = blockIdx.z;
  const int tx0 = blockIdx.x * 16;
  const int ty0 = blockIdx.y * 16;
  const int tid = threadIdx.x;

  const float* srcb = src + (size_t)b * HWSZ;
  const bool interior = (tx0 >= 3) && (tx0 + 18 < WW) && (ty0 >= 3) && (ty0 + 18 < HH);

  if (interior) {
#if __has_builtin(__builtin_amdgcn_tensor_load_to_lds)
    // ---- TDM fill: one wave issues the 2D descriptor, waits TENSORcnt ------
    if (tid < 32) {
      const float* gsrc = srcb + (size_t)(ty0 - 3) * WW + (tx0 - 3);
      unsigned long long ga = (unsigned long long)(uintptr_t)gsrc;
      unsigned int lds_addr = (unsigned int)(uintptr_t)(&sF[0]); // low 32 = LDS offset

      u32x4 g0;
      g0[0] = 1u;                                     // count=1, user descriptor
      g0[1] = lds_addr;                               // lds_addr
      g0[2] = (unsigned int)(ga & 0xFFFFFFFFu);       // global_addr[31:0]
      g0[3] = (unsigned int)((ga >> 32) & 0x1FFFFFFu) // global_addr[56:32]
              | (2u << 30);                           // type = 2 ("image")

      i32x8 g1;
      g1[0] = (2 << 16);          // workgroup_mask=0, data_size=2 (4B)
      g1[1] = (22 << 16);         // tensor_dim0 = 22 (tile-local OOB bound)
      g1[2] = (22 << 16);         // tensor_dim0 hi=0 | tensor_dim1 = 22
      g1[3] = (22 << 16);         // tensor_dim1 hi=0 | tile_dim0 = 22
      g1[4] = 22;                 // tile_dim1 = 22, tile_dim2 = 0
      g1[5] = WW;                 // tensor_dim0_stride = W (row pitch)
      g1[6] = 0;                  // stride0 hi | tensor_dim1_stride lo
      g1[7] = 0;
      i32x4 gz = {0, 0, 0, 0};

#if defined(__clang_major__) && (__clang_major__ >= 23)
      i32x8 gz8 = {0, 0, 0, 0, 0, 0, 0, 0};
      __builtin_amdgcn_tensor_load_to_lds(g0, g1, gz, gz, gz8, 0);
#else
      __builtin_amdgcn_tensor_load_to_lds(g0, g1, gz, gz, 0);
#endif
      __builtin_amdgcn_s_wait_tensorcnt(0);
    }
    __syncthreads();
#else
    for (int i = tid; i < 22 * 22; i += 256) {
      int ly = i / 22, lx = i - ly * 22;
      sF[i] = srcb[(size_t)(ty0 - 3 + ly) * WW + (tx0 - 3 + lx)];
    }
    __syncthreads();
#endif
    prop_rows<false>(gbf, dynm, finit, ffix, fixp, aff, dst, sF, b, tx0, ty0, iter);
  } else {
    // ---- border: zero-padded scalar fill -----------------------------------
    for (int i = tid; i < 22 * 22; i += 256) {
      int ly = i / 22, lx = i - ly * 22;
      int gy = ty0 - 3 + ly;
      int gx = tx0 - 3 + lx;
      float v = 0.f;
      if ((unsigned)gy < (unsigned)HH && (unsigned)gx < (unsigned)WW)
        v = srcb[(size_t)gy * WW + gx];
      sF[i] = v;
    }
    __syncthreads();
    prop_rows<true>(gbf, dynm, finit, ffix, fixp, aff, dst, sF, b, tx0, ty0, iter);
  }
}

// ---------------------------------------------------------------------------
// launch
// ---------------------------------------------------------------------------
extern "C" void kernel_launch(void* const* d_in, const int* in_sizes, int n_in,
                              void* d_out, int out_size, void* d_ws, size_t ws_size,
                              hipStream_t stream) {
  (void)in_sizes; (void)n_in; (void)out_size; (void)ws_size;

  const float* feat_init = (const float*)d_in[0];
  const float* guidance  = (const float*)d_in[1];
  const float* dynamic   = (const float*)d_in[2];
  const float* confid    = (const float*)d_in[3];
  const float* feat_fix  = (const float*)d_in[4];
  float* out = (float*)d_out;

  // workspace layout
  char* ws = (char*)d_ws;
  unsigned short* gbf  = (unsigned short*)ws;                 // 48 bf16 planes
  size_t off = (size_t)BB * CG * HWSZ * sizeof(unsigned short);
  float* aff  = (float*)(ws + off); off += (size_t)6 * PLANE * sizeof(float);
  float* fixp = (float*)(ws + off); off += (size_t)PLANE * sizeof(float);
  float* fA   = (float*)(ws + off); off += (size_t)PLANE * sizeof(float);
  float* fB   = (float*)(ws + off);

  // pass 1: iteration-invariant precompute
  int nblk = (PLANE + 255) / 256;
  precompute_kernel<<<nblk, 256, 0, stream>>>(guidance, confid, feat_fix,
                                              gbf, aff, fixp);

  // passes 2..7: six propagation iterations (ping-pong, final -> d_out)
  dim3 grid(WW / 16, HH / 16, BB);
  for (int k = 0; k < 6; ++k) {
    const float* src = (k == 0) ? feat_init : ((k & 1) ? fA : fB);
    float* dst       = (k == 5) ? out       : ((k & 1) ? fB : fA);
    prop_iter_kernel<<<grid, 256, 0, stream>>>(gbf, dynamic, feat_init,
                                               feat_fix, fixp, aff,
                                               src, dst, k);
  }
}
